// SelfAttention_5626407157803
// MI455X (gfx1250) — matrix operand order repaired and checked by reading, not compiled, not run on hardware
//
#include <hip/hip_runtime.h>
#include <hip/hip_fp16.h>
#include <hip/hip_bf16.h>

typedef __attribute__((ext_vector_type(16))) _Float16 v16h;
typedef __attribute__((ext_vector_type(8)))  float    v8f;

#define WMMA_F16(a, b, c) \
  __builtin_amdgcn_wmma_f32_16x16x32_f16(false, (a), false, (b), (short)0, (c), false, false)

// ---- problem constants ----
#define BB   4
#define CC   512
#define CK   64
#define NN   4096
#define NT   256        // 4096/16 tiles
#define KBX  16         // 512/32 k-blocks for projection

// ---- WMMA 16-bit fragment swizzles (ISA 7.12.2) ----
// A (16x32, row striped within lane): lane = m + 16*k[3]; VGPR r: k = 16*r[2] + 2*r[1:0] + 8*hi + p
__device__ __forceinline__ void frag_pos_a(int k32, int m, int& lane, int& e) {
  lane = m + (((k32 >> 3) & 1) << 4);
  int r = ((k32 >> 1) & 3) | ((k32 >> 4) << 2);
  e = (r << 1) | (k32 & 1);
}
// B (32x16, column striped across lanes): lane = n + 16*k[3]; e = (k & 7) + 8*k[4] (same k order as A)
__device__ __forceinline__ void frag_pos_b(int k32, int n, int& lane, int& e) {
  lane = n + (((k32 >> 3) & 1) << 4);
  e = (k32 & 7) | ((k32 >> 4) << 3);
}

// ---------------- kernel 1: x fp32 -> f16 B-fragment layout ----------------
// xfrag[b][kb(16)][jt(256)][lane(32)][e(16)]
__global__ void sa_convert_x(const float* __restrict__ x, _Float16* __restrict__ xfrag) {
  size_t tid = (size_t)blockIdx.x * 256 + threadIdx.x;   // B*C*N threads
  int n = (int)(tid & (NN - 1));
  int c = (int)((tid >> 12) & (CC - 1));
  int b = (int)(tid >> 21);
  float v = x[tid];
  int kb = c >> 5, k32 = c & 31, jt = n >> 4;
  int l2, e2; frag_pos_b(k32, n & 15, l2, e2);
  xfrag[(((size_t)(b * KBX + kb) * NT + jt) << 9) + l2 * 16 + e2] = (_Float16)v;
}

// ---------------- kernel 2: Wq/Wk fp32 -> f16 A-fragment layout ----------------
// wfrag[mtile(4)][kb(16)][lane(32)][e(16)]
__global__ void sa_convert_w(const float* __restrict__ Wq, const float* __restrict__ Wk,
                             _Float16* __restrict__ wqf, _Float16* __restrict__ wkf) {
  int tid = blockIdx.x * 256 + threadIdx.x;  // 2*64*512
  int which = tid >> 15;
  int r = tid & 32767;
  int krow = r >> 9, c = r & (CC - 1);
  const float* src = which ? Wk : Wq;
  _Float16* dst = which ? wkf : wqf;
  int l2, e2; frag_pos_a(c & 31, krow & 15, l2, e2);
  dst[(((krow >> 4) * KBX + (c >> 5)) << 9) + l2 * 16 + e2] = (_Float16)src[krow * CC + c];
}

// ---------------- kernel 3: zero betabar accumulator ----------------
__global__ void sa_init_bb(float* __restrict__ bb) {
  int i = blockIdx.x * 256 + threadIdx.x;
  if (i < BB * NN) bb[i] = 0.f;
}

// ---------------- kernel 4: projections f = Wq*x, g = Wk*x (WMMA) ----------------
// grid (jt=256, b=4), block 128 (4 waves, wave = mtile). Output: f in scores-A layout,
// g in scores-B layout, both [b][kb2(2)][tile(256)][512] halfs.
__global__ void __launch_bounds__(128) sa_proj(const _Float16* __restrict__ xfrag,
                        const _Float16* __restrict__ wqf, const _Float16* __restrict__ wkf,
                        _Float16* __restrict__ ffrag, _Float16* __restrict__ gfrag) {
  int jt = blockIdx.x, b = blockIdx.y;
  int wave = threadIdx.x >> 5;          // mtile 0..3
  int lane = threadIdx.x & 31;
  v8f accF = {}; v8f accG = {};
  for (int kb = 0; kb < KBX; ++kb) {
    v16h xb = *(const v16h*)(xfrag + (((size_t)(b * KBX + kb) * NT + jt) << 9) + lane * 16);
    v16h aq = *(const v16h*)(wqf + (((wave * KBX + kb) << 9)) + lane * 16);
    v16h ak = *(const v16h*)(wkf + (((wave * KBX + kb) << 9)) + lane * 16);
    accF = WMMA_F16(aq, xb, accF);
    accG = WMMA_F16(ak, xb, accG);
  }
  int col = lane & 15, hi = lane >> 4;
  for (int r = 0; r < 8; ++r) {
    int krow = wave * 16 + r + 8 * hi;      // 0..63 (low-rank dim)
    int kb2 = krow >> 5, k32 = krow & 31;
    size_t base = ((size_t)((b * 2 + kb2) * NT + jt)) << 9;
    int la, ea; frag_pos_a(k32, col, la, ea);
    int lb, eb; frag_pos_b(k32, col, lb, eb);
    ffrag[base + la * 16 + ea] = (_Float16)accF[r];
    gfrag[base + lb * 16 + eb] = (_Float16)accG[r];
  }
}

// ---------------- kernel 5: scores + online softmax + beta + row-sums ----------------
// grid (64, 4), block 128. WG covers 4 j-tiles (64 columns); each wave computes ALL 4
// j-tiles over 1/4 of the i-range (cuts f re-reads from L2 by 4x, 8 WMMAs per A-fetch).
// Online-softmax stats merged across the 4 i-quarters via LDS between passes.
__global__ void __launch_bounds__(128) sa_scores(const _Float16* __restrict__ ffrag,
                          const _Float16* __restrict__ gfrag,
                          float* __restrict__ beta, float* __restrict__ betabar) {
  __shared__ float smax[4][4][32];   // [wave][tile][lane]
  __shared__ float ssum[4][4][32];
  int b = blockIdx.y;
  int w = threadIdx.x >> 5;
  int lane = threadIdx.x & 31;
  int jt0 = blockIdx.x * 4;

  // hoist B (g) fragments for the 4 column tiles: K=64 -> 2 frags each (64 VGPRs)
  v16h g[4][2];
#pragma unroll
  for (int t = 0; t < 4; ++t)
#pragma unroll
    for (int k = 0; k < 2; ++k)
      g[t][k] = *(const v16h*)(gfrag + (((size_t)((b * 2 + k) * NT + (jt0 + t))) << 9) + lane * 16);

  float m[4], s[4];
#pragma unroll
  for (int t = 0; t < 4; ++t) { m[t] = -INFINITY; s[t] = 0.f; }

  int itBeg = w * (NT / 4), itEnd = itBeg + (NT / 4);
  // ---- pass 1: online max/sum over this wave's i-quarter, per column ----
  for (int it = itBeg; it < itEnd; ++it) {
    v16h a0 = *(const v16h*)(ffrag + (((size_t)((b * 2 + 0) * NT + it)) << 9) + lane * 16);
    v16h a1 = *(const v16h*)(ffrag + (((size_t)((b * 2 + 1) * NT + it)) << 9) + lane * 16);
#pragma unroll
    for (int t = 0; t < 4; ++t) {
      v8f acc = {};
      acc = WMMA_F16(a0, g[t][0], acc);
      acc = WMMA_F16(a1, g[t][1], acc);
      float tmax = acc[0];
#pragma unroll
      for (int r = 1; r < 8; ++r) tmax = fmaxf(tmax, acc[r]);
      tmax = fmaxf(tmax, __shfl_xor(tmax, 16, 32));   // combine the two row-halves
      float mn = fmaxf(m[t], tmax);
      float ps = 0.f;
#pragma unroll
      for (int r = 0; r < 8; ++r) ps += __expf(acc[r] - mn);
      ps += __shfl_xor(ps, 16, 32);
      s[t] = s[t] * __expf(m[t] - mn) + ps;
      m[t] = mn;
    }
  }

  // ---- merge online-softmax stats across the 4 i-quarters ----
#pragma unroll
  for (int t = 0; t < 4; ++t) { smax[w][t][lane] = m[t]; ssum[w][t][lane] = s[t]; }
  __syncthreads();
  float inv[4];
#pragma unroll
  for (int t = 0; t < 4; ++t) {
    float mm = smax[0][t][lane];
#pragma unroll
    for (int ww = 1; ww < 4; ++ww) mm = fmaxf(mm, smax[ww][t][lane]);
    float st = 0.f;
#pragma unroll
    for (int ww = 0; ww < 4; ++ww) st += ssum[ww][t][lane] * __expf(smax[ww][t][lane] - mm);
    m[t] = mm;
    inv[t] = 1.0f / st;
  }

  int col = lane & 15, hi = lane >> 4;
  // ---- pass 2: recompute, normalize, write beta, accumulate row-sums ----
  for (int it = itBeg; it < itEnd; ++it) {
    v16h a0 = *(const v16h*)(ffrag + (((size_t)((b * 2 + 0) * NT + it)) << 9) + lane * 16);
    v16h a1 = *(const v16h*)(ffrag + (((size_t)((b * 2 + 1) * NT + it)) << 9) + lane * 16);
    v8f acc[4];
#pragma unroll
    for (int t = 0; t < 4; ++t) {
      v8f a = {};
      a = WMMA_F16(a0, g[t][0], a);
      a = WMMA_F16(a1, g[t][1], a);
      acc[t] = a;
    }
#pragma unroll
    for (int r = 0; r < 8; ++r) {
      int i = it * 16 + r + 8 * hi;
      float wsum = 0.f;
#pragma unroll
      for (int t = 0; t < 4; ++t) {
        float v = __expf(acc[t][r] - m[t]) * inv[t];
        beta[((size_t)(b * NN + i) << 12) + (jt0 + t) * 16 + col] = v;  // coalesced over lanes 0-15
        wsum += v;
      }
      wsum += __shfl_xor(wsum, 1, 32);   // row-sum over the WG's 64 columns
      wsum += __shfl_xor(wsum, 2, 32);
      wsum += __shfl_xor(wsum, 4, 32);
      wsum += __shfl_xor(wsum, 8, 32);
      if (col == 0) atomicAdd(&betabar[b * NN + i], wsum);
    }
  }
}

// ---------------- kernel 6: t[b,d] = x[b,d,:]·betabar/N ; xmean ----------------
// grid (64, 4), block 256: one wave per row d.
__global__ void sa_gemv(const float* __restrict__ x, const float* __restrict__ bb,
                        float* __restrict__ t, float* __restrict__ xmean) {
  int b = blockIdx.y;
  int d = blockIdx.x * 8 + (threadIdx.x >> 5);
  int lane = threadIdx.x & 31;
  const float* row = x + ((size_t)(b * CC + d) << 12);
  const float* bbp = bb + b * NN;
  float accT = 0.f, accX = 0.f;
  for (int i = lane; i < NN; i += 32) {
    float xv = row[i];
    accT = fmaf(xv, bbp[i], accT);
    accX += xv;
  }
#pragma unroll
  for (int msk = 16; msk >= 1; msk >>= 1) {
    accT += __shfl_xor(accT, msk, 32);
    accX += __shfl_xor(accX, msk, 32);
  }
  if (lane == 0) {
    t[b * CC + d]     = accT * (1.0f / NN);
    xmean[b * CC + d] = accX * (1.0f / NN);
  }
}

// ---------------- kernel 7: pooled[b,c] = gamma*Wv[c,:]·t[b,:] + xmean ----------------
__global__ void sa_pooled(const float* __restrict__ Wv, const float* __restrict__ t,
                          const float* __restrict__ xmean, const float* __restrict__ gamma,
                          float* __restrict__ pooled) {
  int idx = blockIdx.x * 8 + (threadIdx.x >> 5);   // 0..2047
  int b = idx >> 9, c = idx & (CC - 1);
  int lane = threadIdx.x & 31;
  const float* wr = Wv + (size_t)c * CC;
  const float* tb = t + b * CC;
  float a = 0.f;
  for (int d = lane; d < CC; d += 32) a = fmaf(wr[d], tb[d], a);
#pragma unroll
  for (int msk = 16; msk >= 1; msk >>= 1) a += __shfl_xor(a, msk, 32);
  if (lane == 0) pooled[idx] = gamma[0] * a + xmean[idx];
}

extern "C" void kernel_launch(void* const* d_in, const int* in_sizes, int n_in,
                              void* d_out, int out_size, void* d_ws, size_t ws_size,
                              hipStream_t stream) {
  const float* x     = (const float*)d_in[0];
  const float* Wq    = (const float*)d_in[1];
  const float* Wk    = (const float*)d_in[2];
  const float* Wv    = (const float*)d_in[3];
  const float* gamma = (const float*)d_in[4];

  float* pooled = (float*)d_out;                 // 4*512
  float* beta   = (float*)d_out + BB * CC;       // 4*4096*4096

  // workspace carving (256B aligned)
  char* ws = (char*)d_ws;
  _Float16* xfrag = (_Float16*)(ws);                        // 16 MB
  _Float16* wqf   = (_Float16*)(ws + 16777216);             // 64 KB
  _Float16* wkf   = (_Float16*)(ws + 16842752);             // 64 KB
  _Float16* ffrag = (_Float16*)(ws + 16908288);             // 2 MB
  _Float16* gfrag = (_Float16*)(ws + 19005440);             // 2 MB
  float*    bbar  = (float*)   (ws + 21102592);             // 64 KB
  float*    tvec  = (float*)   (ws + 21168128);             // 8 KB
  float*    xmean = (float*)   (ws + 21176320);             // 8 KB

  sa_convert_x<<<dim3((BB * CC * NN) / 256), dim3(256), 0, stream>>>(x, xfrag);
  sa_convert_w<<<dim3(256), dim3(256), 0, stream>>>(Wq, Wk, wqf, wkf);
  sa_init_bb<<<dim3(64), dim3(256), 0, stream>>>(bbar);
  sa_proj<<<dim3(NT, BB), dim3(128), 0, stream>>>(xfrag, wqf, wkf, ffrag, gfrag);
  sa_scores<<<dim3(NT / 4, BB), dim3(128), 0, stream>>>(ffrag, gfrag, beta, bbar);
  sa_gemv<<<dim3(CC / 8, BB), dim3(256), 0, stream>>>(x, bbar, tvec, xmean);
  sa_pooled<<<dim3((BB * CC) / 8), dim3(256), 0, stream>>>(Wv, tvec, xmean, gamma, pooled);
}
